// DMENet_55456617725984
// MI455X (gfx1250) — matrix operand order repaired and checked by reading, not compiled, not run on hardware
//
#include <hip/hip_runtime.h>
#include <hip/hip_bf16.h>

// ---------------------------------------------------------------------------
// DMENet forward on gfx1250 (MI455X), NHWC edition.
//
// Activations NHWC: act[p][c], p = y*W+x, channel dim padded to multiple of
// 32 (only the network input needs real padding, 3 -> 32).  Every contraction
// runs through V_WMMA_F32_16X16X32_F16 implicit GEMM, decomposed tap-wise:
//     out += sum_tap W_tap[Cout][Cin] x shift(in,tap)[Cin][HW]
//
// B fragment per lane = 16 consecutive channels of one pixel: one 32B-aligned
// v16h load.  D epilogue per lane = 8 consecutive output channels of one
// pixel: one 16B v8h store (branch concat = channel offset + stride).
// Weights pre-transposed to [tap][Coutpad][Cinpad] f16, staged through LDS in
// 256-element chunks.  All spatial dims are powers of two -> shift/mask
// addressing (no integer division in hot kernels).
//
// Block = 128 threads (4 wave32): one 16-row Cout tile x 256 pixels; each
// wave computes 4 N-tiles sharing one A fragment; K-loop unrolled x2.
// ---------------------------------------------------------------------------

typedef __attribute__((ext_vector_type(16))) _Float16 v16h;
typedef __attribute__((ext_vector_type(8)))  _Float16 v8h;
typedef __attribute__((ext_vector_type(8)))  float    v8f;

#define KCHUNK 256   // halves of the channel (reduction) dim staged in LDS

// ---------------- helper kernels -------------------------------------------

// x NCHW [3][HW] f32 -> NHWC [HW][32] f16, channels 3..31 zeroed.
__global__ void cvt_in_kernel(const float* __restrict__ src,
                              _Float16* __restrict__ dst, int HW) {
    int i = blockIdx.x * blockDim.x + threadIdx.x;
    if (i >= HW * 32) return;
    int c = i & 31;
    int p = i >> 5;
    dst[i] = (c < 3) ? (_Float16)src[(size_t)c * HW + p] : (_Float16)0.f;
}

// Weights (Cout, Cin, ksz, ksz) f32 -> [tap][Coutpad][Cinpad] f16, zero pad.
__global__ void cvt_wgt_tap_kernel(const float* __restrict__ src,
                                   _Float16* __restrict__ dst,
                                   int Cout, int Cin, int KK,
                                   int Coutpad, int Cinpad) {
    size_t i = (size_t)blockIdx.x * blockDim.x + threadIdx.x;
    size_t tot = (size_t)KK * Coutpad * Cinpad;
    if (i >= tot) return;
    int cc = (int)(i % Cinpad);
    size_t t2 = i / Cinpad;
    int o   = (int)(t2 % Coutpad);
    int tap = (int)(t2 / Coutpad);
    float v = (o < Cout && cc < Cin) ? src[((size_t)o * Cin + cc) * KK + tap] : 0.0f;
    dst[i] = (_Float16)v;
}

// 2x2 maxpool, NHWC.  C and W are powers of two.
__global__ void maxpool_kernel(const _Float16* __restrict__ in,
                               _Float16* __restrict__ out,
                               int logC, int logW, int total) {
    int i = blockIdx.x * blockDim.x + threadIdx.x;
    if (i >= total) return;
    const int C = 1 << logC, W = 1 << logW;
    int c  = i & (C - 1);
    int po = i >> logC;
    int xo = po & ((W >> 1) - 1);
    int yo = po >> (logW - 1);
    const _Float16* p0 = in + ((size_t)((yo << 1) << logW) + (xo << 1)) * C + c;
    float a = fmaxf((float)p0[0], (float)p0[C]);
    float b = fmaxf((float)p0[(size_t)W * C], (float)p0[(size_t)W * C + C]);
    out[i] = (_Float16)fmaxf(a, b);
}

// DCNv2 bilinear sampling (NHWC in / NHWC out):
//   samp[p][c*KK+j] = bilerp(x[.,c], p + koff_j + off_j(p)) * mask_j(p)
__global__ void deform_sample_kernel(const _Float16* __restrict__ in,   // [HW][Kc]
                                     const float* __restrict__ off,     // [2*KK][HW]
                                     const float* __restrict__ mask,    // [KK][HW]
                                     _Float16* __restrict__ samp,       // [HW][C*KK]
                                     int C, int Kc, int logW, int H,
                                     int ksz, int pad) {
    const int W  = 1 << logW;
    const int HW = H << logW;
    const int KK = ksz * ksz;
    const int CKK = C * KK;
    size_t total = (size_t)HW * CKK;
    size_t i = (size_t)blockIdx.x * blockDim.x + threadIdx.x;
    if (i >= total) return;
    int ck = (int)(i % CKK);
    int p  = (int)(i / CKK);
    int j = ck % KK;
    int c = ck / KK;
    int y = p >> logW, x = p & (W - 1);
    float py = (float)(y + j / ksz - pad) + off[(size_t)(2 * j) * HW + p];
    float px = (float)(x + j % ksz - pad) + off[(size_t)(2 * j + 1) * HW + p];
    float y0f = floorf(py), x0f = floorf(px);
    float dy = py - y0f, dx = px - x0f;
    int y0 = (int)y0f, x0 = (int)x0f;
    auto g = [&](int yi, int xi) -> float {
        if ((unsigned)yi >= (unsigned)H || (unsigned)xi >= (unsigned)W) return 0.f;
        return (float)in[((size_t)(yi << logW) + xi) * Kc + c];
    };
    float v = g(y0, x0)         * (1.f - dy) * (1.f - dx)
            + g(y0, x0 + 1)     * (1.f - dy) * dx
            + g(y0 + 1, x0)     * dy * (1.f - dx)
            + g(y0 + 1, x0 + 1) * dy * dx;
    samp[i] = (_Float16)(v * mask[(size_t)j * HW + p]);
}

// ---------------- WMMA implicit-GEMM conv (tap-wise, NHWC) ------------------
// grid = (ceil(HW/256), Coutpad/16), block = 128 (4 waves).
// act: 0 = none, 1 = relu, 2 = sigmoid.
// out16: NHWC, per-pixel stride `ostride` (Cout multiple of 16 required).
// out32: NCHW [Cout][HW] scalar path (offset/mask convs, final output).

__global__ void __launch_bounds__(128)
conv_wmma_kernel(const _Float16* __restrict__ in,    // NHWC [HW][Kc]
                 const _Float16* __restrict__ wgt,   // [KK][Coutpad][Kc] f16
                 const float* __restrict__ bias,     // [Cout] or null
                 _Float16* __restrict__ out16,       // NHWC or null
                 float* __restrict__ out32,          // NCHW or null
                 int Kc, int logW, int H, int Cout,
                 int ksz, int pad, int Coutpad, int ostride, int act) {
    __shared__ uint32_t aLds[16 * (KCHUNK / 2)];     // 16 rows x 256 halves = 8KB
    const _Float16* aH = (const _Float16*)aLds;

    const int W    = 1 << logW;
    const int HW   = H << logW;
    const int tid  = threadIdx.x;
    const int wave = tid >> 5;
    const int lane = tid & 31;
    const int hi   = lane >> 4;
    const int l16  = lane & 15;
    const int m0   = blockIdx.y * 16;

    // Four 16-pixel N-tiles per wave; every 16-pixel group lies in one image
    // row (W is always a multiple of 16 here).
    const int pixbase = blockIdx.x * 256 + wave * 64;
    int  ty[4], tx[4], tp[4];
    bool tval[4];
#pragma unroll
    for (int t = 0; t < 4; ++t) {
        int p0 = pixbase + t * 16;
        tval[t] = (p0 < HW);
        int pc  = tval[t] ? p0 : 0;
        ty[t]   = pc >> logW;
        tx[t]   = (pc & (W - 1)) + l16;
        tp[t]   = p0 + l16;
    }

    v8f acc[4] = {};

    const _Float16* wtap = wgt;                       // advances per tap
    const size_t wtapStep = (size_t)Coutpad * Kc;

    for (int tky = 0; tky < ksz; ++tky) {
        const int dy = tky - pad;
        for (int tkx = 0; tkx < ksz; ++tkx, wtap += wtapStep) {
            const int dx = tkx - pad;

            // per-tile shifted pixel pointer + validity (constant over K loop)
            const _Float16* bp[4];
            bool bval[4];
#pragma unroll
            for (int t = 0; t < 4; ++t) {
                int iy = ty[t] + dy;
                int ix = tx[t] + dx;
                bval[t] = tval[t] && ((unsigned)iy < (unsigned)H) && ((unsigned)ix < (unsigned)W);
                size_t pix = bval[t] ? (((size_t)iy << logW) + ix) : 0;
                bp[t] = in + pix * Kc;
            }

            for (int c0 = 0; c0 < Kc; c0 += KCHUNK) {
                const int chunk = (Kc - c0 < KCHUNK) ? (Kc - c0) : KCHUNK;  // mult. 32
                const int u32PerRow = chunk >> 1;
#pragma unroll 4
                for (int j = tid; j < 16 * (KCHUNK / 2); j += 128) {
                    int row = j >> 7;          // KCHUNK/2 == 128
                    int col = j & 127;
                    uint32_t v = 0;
                    if (col < u32PerRow)
                        v = *(const uint32_t*)(wtap + (size_t)(m0 + row) * Kc + c0 + col * 2);
                    aLds[(row << 7) + col] = v;
                }
                if (c0 + KCHUNK < Kc)          // pre-touch next weight chunk
                    __builtin_prefetch(wtap + (size_t)(m0 + l16) * Kc + c0 + KCHUNK, 0, 3);
                __syncthreads();

#pragma unroll 2
                for (int cc = 0; cc < chunk; cc += 32) {
                    // ---- A fragment (16x32 f16, contiguous k = 16*hi + s) from LDS
                    const _Float16* arow = aH + l16 * KCHUNK + cc;
                    v16h a;
#pragma unroll
                    for (int i = 0; i < 4; ++i) {
                        int kk = hi * 16 + 2 * i;
                        a[2 * i]         = arow[kk];
                        a[2 * i + 1]     = arow[kk + 1];
                        a[8 + 2 * i]     = arow[8 + kk];
                        a[8 + 2 * i + 1] = arow[8 + kk + 1];
                    }
                    // ---- B fragments: one 32B-aligned v16h load per tile --
                    const int ch0 = c0 + cc + hi * 16;   // multiple of 16
                    v16h b[4];
#pragma unroll
                    for (int t = 0; t < 4; ++t) {
                        v16h bb = {};
                        if (bval[t]) bb = *(const v16h*)(bp[t] + ch0);
                        b[t] = bb;
                    }
                    // ---- 4 WMMAs sharing one A fragment -------------------
#pragma unroll
                    for (int t = 0; t < 4; ++t)
                        acc[t] = __builtin_amdgcn_wmma_f32_16x16x32_f16(
                            false, a, false, b[t], (short)0, acc[t], false, false);
                }
                __syncthreads();
            }
        }
    }

    // ---- epilogue: D f32 16x16 (VGPR i -> M = i (+8 for hi half)) ---------
    const int mbase = m0 + hi * 8;     // 8 consecutive output channels per lane
#pragma unroll
    for (int t = 0; t < 4; ++t) {
        if (!tval[t]) continue;
        int p = tp[t];
        float v[8];
#pragma unroll
        for (int i = 0; i < 8; ++i) {
            float u = acc[t][i] + (bias ? bias[mbase + i] : 0.0f);
            if (act == 1)      u = fmaxf(u, 0.0f);
            else if (act == 2) u = 1.0f / (1.0f + __expf(-u));
            v[i] = u;
        }
        if (out16 && mbase + 8 <= Cout) {          // Cout mult. of 16 on this path
            v8h h;
#pragma unroll
            for (int i = 0; i < 8; ++i) h[i] = (_Float16)v[i];
            *(v8h*)(out16 + (size_t)p * ostride + mbase) = h;
        }
        if (out32) {
#pragma unroll
            for (int i = 0; i < 8; ++i)
                if (mbase + i < Cout) out32[(size_t)(mbase + i) * HW + p] = v[i];
        }
    }
}

// ---------------- host orchestration ---------------------------------------

static inline int ilog2(int v) { return 31 - __builtin_clz((unsigned)v); }

extern "C" void kernel_launch(void* const* d_in, const int* in_sizes, int n_in,
                              void* d_out, int out_size, void* d_ws, size_t ws_size,
                              hipStream_t stream) {
    (void)in_sizes; (void)n_in; (void)out_size; (void)ws_size;

    size_t woff = 0;
    auto alloc = [&](size_t bytes) -> void* {
        woff = (woff + 255) & ~(size_t)255;
        void* p = (char*)d_ws + woff;
        woff += bytes;
        return p;
    };
    _Float16* bufA    = (_Float16*)alloc((size_t)64 * 256 * 256 * sizeof(_Float16));
    _Float16* bufB    = (_Float16*)alloc((size_t)64 * 256 * 256 * sizeof(_Float16));
    _Float16* wbuf    = (_Float16*)alloc((size_t)256 * 25088 * sizeof(_Float16));
    float*    offbuf  = (float*)   alloc((size_t)2 * 49 * 1024 * sizeof(float));
    float*    maskbuf = (float*)   alloc((size_t)49 * 1024 * sizeof(float));
    _Float16* sampbuf = (_Float16*)alloc((size_t)25088 * 1024 * sizeof(_Float16));
    _Float16* concatb = (_Float16*)alloc((size_t)768 * 1024 * sizeof(_Float16));

    // in: NHWC with Kc = ceil32(Cin) channels per pixel.
    auto conv = [&](const _Float16* in, int Cin, int H, int W,
                    const float* w32, const float* b32, int Cout,
                    int ksz, int pad, int act,
                    _Float16* o16, int ostride, float* o32) {
        int KK  = ksz * ksz;
        int Kc  = (Cin + 31) & ~31;
        int Cop = (Cout + 15) & ~15;
        size_t wtot = (size_t)KK * Cop * Kc;
        cvt_wgt_tap_kernel<<<dim3((unsigned)((wtot + 255) / 256)), 256, 0, stream>>>(
            w32, wbuf, Cout, Cin, KK, Cop, Kc);
        dim3 grid((unsigned)((H * W + 255) / 256), (unsigned)(Cop / 16));
        conv_wmma_kernel<<<grid, 128, 0, stream>>>(
            in, wbuf, b32, o16, o32, Kc, ilog2(W), H, Cout, ksz, pad, Cop, ostride, act);
    };

    // ---- input -> NHWC f16, channel-padded 3 -> 32 ----
    const float* x = (const float*)d_in[0];
    int C = 3, H = 256, W = 256;
    cvt_in_kernel<<<(32 * 256 * 256 + 255) / 256, 256, 0, stream>>>(x, bufA, 256 * 256);
    _Float16* cur = bufA; _Float16* nxt = bufB;

    // ---- VGG backbone ----
    const int vggC[10] = {64, 64, 128, 128, 256, 256, 256, 512, 512, 512};
    int idx = 1;
    for (int i = 0; i < 10; ++i) {
        int Co = vggC[i];
        conv(cur, C, H, W, (const float*)d_in[idx], (const float*)d_in[idx + 1],
             Co, 3, 1, /*relu*/1, nxt, Co, nullptr);
        idx += 2;
        _Float16* t = cur; cur = nxt; nxt = t; C = Co;
        if (i == 1 || i == 3 || i == 6) {
            int total = C * (H / 2) * (W / 2);
            maxpool_kernel<<<(total + 255) / 256, 256, 0, stream>>>(
                cur, nxt, ilog2(C), ilog2(W), total);
            t = cur; cur = nxt; nxt = t;
            H >>= 1; W >>= 1;
        }
    }

    // ---- DCNv2 blocks (H = W = 32) ----
    const int cins[3]  = {512, 256, 128};
    const int couts[3] = {256, 128, 64};
    const int pouts[3] = {256, 128, 1};
    for (int blk = 0; blk < 3; ++blk) {
        int cin = cins[blk], cout = couts[blk];
        int ccat = 3 * cout;          // concat channel count (768/384/192)
        int coff = 0;
        for (int br = 0; br < 3; ++br) {
            int ksz = 3 + 2 * br, KK = ksz * ksz, pad = (ksz - 1) / 2;
            const float* ow = (const float*)d_in[idx + 0];
            const float* ob = (const float*)d_in[idx + 1];
            const float* mw = (const float*)d_in[idx + 2];
            const float* mb = (const float*)d_in[idx + 3];
            const float* dw = (const float*)d_in[idx + 4];
            idx += 5;
            conv(cur, cin, H, W, ow, ob, 2 * KK, ksz, pad, 0, nullptr, 0, offbuf);
            conv(cur, cin, H, W, mw, mb, KK,     ksz, pad, 2, nullptr, 0, maskbuf);
            size_t stotal = (size_t)cin * KK * H * W;
            deform_sample_kernel<<<dim3((unsigned)((stotal + 255) / 256)), 256, 0, stream>>>(
                cur, offbuf, maskbuf, sampbuf, cin, cin /*Kc==cin*/, ilog2(W), H, ksz, pad);
            // channel contraction == 1x1 conv over Cin = cin*KK (mult. of 32),
            // written straight into the concat slot (NHWC stride = ccat).
            conv(sampbuf, cin * KK, H, W, dw, nullptr, cout, 1, 0, 0,
                 concatb + coff, ccat, nullptr);
            coff += cout;
        }
        const float* pw = (const float*)d_in[idx + 0];
        const float* pb = (const float*)d_in[idx + 1];
        idx += 2;
        if (blk < 2) {
            conv(concatb, ccat, H, W, pw, pb, pouts[blk], 1, 0, 0, nxt, pouts[blk], nullptr);
            _Float16* t = cur; cur = nxt; nxt = t; C = pouts[blk];
        } else {
            conv(concatb, ccat, H, W, pw, pb, pouts[blk], 1, 0, 0, nullptr, 0, (float*)d_out);
        }
    }
}
